// Enformer_83296595738777
// MI455X (gfx1250) — compile-verified, matrix operand order checked by reading, not run on hardware
//
#include <hip/hip_runtime.h>
#include <math.h>

// ---------------------------------------------------------------------------
// Types for CDNA5 WMMA (gfx1250): D = A(16x32 bf16) * B(32x16 bf16) + C(f32)
// ---------------------------------------------------------------------------
typedef __attribute__((ext_vector_type(16))) __bf16 v16bf;
typedef __attribute__((ext_vector_type(8)))  float  v8f;
typedef __attribute__((ext_vector_type(4)))  unsigned int u32x4;
typedef __attribute__((ext_vector_type(8)))  unsigned int u32x8;

union FragBF {
  v16bf bf;
  unsigned int u32[8];
};

__device__ __forceinline__ unsigned short f2bf(float f) {
  unsigned int u = __float_as_uint(f);
  u += 0x7FFFu + ((u >> 16) & 1u);   // round-to-nearest-even
  return (unsigned short)(u >> 16);
}

__device__ __forceinline__ float gelu_exact(float x) {
  return 0.5f * x * (1.0f + erff(x * 0.70710678118654752440f));
}

// ---------------------------------------------------------------------------
// WMMA GEMM: C[M,N](f32) (+)= A[M,K](bf16,row-major) * B[K,N](bf16,row-major)
// Block tile 64x64, 8 wave32 waves (4 along M x 2 along N), wave tile 16x32.
// A-tile is fetched by the Tensor Data Mover (2D tile, OOB auto-zero-fill);
// B-tile is staged transposed into LDS with branch-free clamped loads.
// ---------------------------------------------------------------------------
#define BM 64
#define BN 64
#define KT 32

__global__ __launch_bounds__(256) void k_gemm_wmma(
    const unsigned short* __restrict__ A, int lda,
    const unsigned short* __restrict__ B, int ldb,
    float* __restrict__ C, int ldc,
    int M, int N, int K, int accum)
{
  __shared__ unsigned short As[BM][KT];   // row-major [m][k], written by TDM
  __shared__ unsigned short Bs[BN][KT];   // transposed [n][k]

  const int tid  = threadIdx.x;
  const int lane = tid & 31;
  const int wave = tid >> 5;
  const int wm   = wave & 3;    // wave row (M)
  const int wn   = wave >> 2;   // wave col (N)
  const int m0   = blockIdx.y * BM;
  const int n0   = blockIdx.x * BN;
  const int fmn  = lane & 15;   // M (for A) / N (for B,C) index within tile
  const int hf   = lane >> 4;   // lane half selects K/M groups per ISA layout

  v8f acc0 = {0.f,0.f,0.f,0.f,0.f,0.f,0.f,0.f};
  v8f acc1 = {0.f,0.f,0.f,0.f,0.f,0.f,0.f,0.f};

  const int bk = tid >> 3;         // B: 32 k-rows, 8 threads/row
  const int bc = (tid & 7) * 8;    //    8 bf16 each (contiguous in N)
  const bool bn_full = (n0 + BN <= N);
  const unsigned int lds_off = (unsigned int)(size_t)(void*)&As[0][0];

  for (int k0 = 0; k0 < K; k0 += KT) {
    // ---- A tile: Tensor Data Mover, issued by wave 0 (uniform descriptor) ----
    if (wave == 0) {
      unsigned long long ga = (unsigned long long)(A + (size_t)m0 * lda + (size_t)k0);
      unsigned int remK = (unsigned int)(K - k0);
      unsigned int remM = (unsigned int)(M - m0);
      u32x4 g0;
      g0[0] = 1u;                                   // count=1, user descriptor
      g0[1] = lds_off;                              // LDS byte address
      g0[2] = (unsigned int)ga;                     // global_addr[31:0]
      g0[3] = (unsigned int)((ga >> 32) & 0x01FFFFFFull) | (2u << 30); // addr[56:32], type=2
      u32x8 g1;
      g1[0] = 0x00010000u;                          // data_size=1 (2 bytes), no flags
      g1[1] = (remK & 0xFFFFu) << 16;               // tensor_dim0[15:0]
      g1[2] = (remK >> 16) | ((remM & 0xFFFFu) << 16);  // dim0[31:16] | dim1[15:0]
      g1[3] = (remM >> 16) | (32u << 16);           // dim1[31:16] | tile_dim0=32
      g1[4] = 64u;                                  // tile_dim1=64, tile_dim2=0
      g1[5] = (unsigned int)lda;                    // tensor_dim0_stride[31:0]
      g1[6] = 0u;                                   // stride hi, dim1_stride lo
      g1[7] = 0u;
      asm volatile("tensor_load_to_lds %0, %1" :: "s"(g0), "s"(g1) : "memory");
    }

    // ---- B tile: stage transposed, branch-free ----
    {
      int gk = k0 + bk;
      int gkc = (gk < K) ? gk : (K - 1);
      const unsigned short* bp = B + (size_t)gkc * ldb + (size_t)(n0 + bc);
      unsigned short tmp[8];
      if ((gk < K) && bn_full && (((size_t)bp & 3) == 0)) {
        const unsigned int* bp32 = (const unsigned int*)bp;
        #pragma unroll
        for (int j = 0; j < 4; ++j) {
          unsigned int w = bp32[j];
          tmp[2 * j]     = (unsigned short)(w & 0xFFFFu);
          tmp[2 * j + 1] = (unsigned short)(w >> 16);
        }
      } else {
        #pragma unroll
        for (int j = 0; j < 8; ++j) {
          int gn = n0 + bc + j;
          int gnc = (gn < N) ? gn : (N - 1);
          unsigned short v = B[(size_t)gkc * ldb + gnc];  // clamped, always in-bounds
          tmp[j] = ((gk < K) && (gn < N)) ? v : (unsigned short)0;
        }
      }
      #pragma unroll
      for (int j = 0; j < 8; ++j) Bs[bc + j][bk] = tmp[j];
    }
    if (k0 + KT < K) { // speculative prefetch of next B K-slice
      int pk = k0 + KT + bk; if (pk >= K) pk = K - 1;
      __builtin_prefetch(B + (size_t)pk * ldb + (size_t)n0, 0, 0);
    }
    if (wave == 0) __builtin_amdgcn_s_wait_tensorcnt(0);
    __syncthreads();

    // Gather fragments per ISA 16-bit layout:
    //   VGPR i holds K pair { base, base+1 }, base = 2*(i&3) + 16*(i>>2) + 8*half
    FragBF fa, fb0, fb1;
    #pragma unroll
    for (int i = 0; i < 8; ++i) {
      int kk = 2 * (i & 3) + 16 * (i >> 2) + 8 * hf;
      fa.u32[i]  = *(const unsigned int*)&As[wm * 16 + fmn][kk];
      fb0.u32[i] = *(const unsigned int*)&Bs[wn * 32 + fmn][kk];
      fb1.u32[i] = *(const unsigned int*)&Bs[wn * 32 + 16 + fmn][kk];
    }
    acc0 = __builtin_amdgcn_wmma_f32_16x16x32_bf16(false, fa.bf, false, fb0.bf,
                                                   (short)0, acc0, false, false);
    acc1 = __builtin_amdgcn_wmma_f32_16x16x32_bf16(false, fa.bf, false, fb1.bf,
                                                   (short)0, acc1, false, false);
    __syncthreads();
  }

  // C/D layout: VGPR r, lane half hf -> M = r + 8*hf ; N = lane&15
  int gn0 = n0 + wn * 32 + fmn;
  int gn1 = gn0 + 16;
  #pragma unroll
  for (int r = 0; r < 8; ++r) {
    int gm = m0 + wm * 16 + r + 8 * hf;
    if (gm < M) {
      if (gn0 < N) {
        size_t idx = (size_t)gm * ldc + gn0;
        float v = acc0[r]; if (accum) v += C[idx]; C[idx] = v;
      }
      if (gn1 < N) {
        size_t idx = (size_t)gm * ldc + gn1;
        float v = acc1[r]; if (accum) v += C[idx]; C[idx] = v;
      }
    }
  }
}

// ---------------------------------------------------------------------------
// Support kernels
// ---------------------------------------------------------------------------
__global__ void k_stem_conv15(const float* __restrict__ x, const float* __restrict__ w,
                              const float* __restrict__ b, float* __restrict__ y, int L)
{
  size_t i = (size_t)blockIdx.x * 256 + threadIdx.x;
  if (i >= (size_t)768 * L) return;
  int co = (int)(i / L), l = (int)(i % L);
  float acc = b[co];
  #pragma unroll
  for (int ci = 0; ci < 4; ++ci) {
    const float* xr = x + (size_t)ci * L;
    const float* wr = w + ((size_t)co * 4 + ci) * 15;
    #pragma unroll
    for (int t = 0; t < 15; ++t) {
      int ll = l + t - 7;
      if (ll >= 0 && ll < L) acc += wr[t] * xr[ll];
    }
  }
  y[i] = acc;
}

__global__ void k_bn_stats(const float* __restrict__ x, int L,
                           float* __restrict__ mean, float* __restrict__ rvar)
{
  int c = blockIdx.x;
  __shared__ float s1[256], s2[256];
  float a = 0.f, q = 0.f;
  for (int l = threadIdx.x; l < L; l += 256) {
    float v = x[(size_t)c * L + l]; a += v; q += v * v;
  }
  s1[threadIdx.x] = a; s2[threadIdx.x] = q; __syncthreads();
  for (int o = 128; o > 0; o >>= 1) {
    if (threadIdx.x < o) { s1[threadIdx.x] += s1[threadIdx.x + o]; s2[threadIdx.x] += s2[threadIdx.x + o]; }
    __syncthreads();
  }
  if (threadIdx.x == 0) {
    float m = s1[0] / (float)L;
    float v = s2[0] / (float)L - m * m;
    mean[c] = m; rvar[c] = rsqrtf(v + 1e-5f);
  }
}

// BN -> GELU -> bf16 with `pad` zero columns on each side (for conv GEMMs)
__global__ void k_bn_gelu_pad(const float* __restrict__ x, const float* __restrict__ mean,
                              const float* __restrict__ rvar, const float* __restrict__ g,
                              const float* __restrict__ b, unsigned short* __restrict__ y,
                              int L, int LP, int pad, size_t total)
{
  size_t i = (size_t)blockIdx.x * 256 + threadIdx.x;
  if (i >= total) return;
  int c = (int)(i / LP), p = (int)(i % LP), l = p - pad;
  unsigned short out = 0;
  if (l >= 0 && l < L) {
    float v = (x[(size_t)c * L + l] - mean[c]) * rvar[c] * g[c] + b[c];
    out = f2bf(gelu_exact(v));
  }
  y[i] = out;
}

__global__ void k_cvt_pad(const float* __restrict__ x, unsigned short* __restrict__ y,
                          int L, int LP, int pad, size_t total)
{
  size_t i = (size_t)blockIdx.x * 256 + threadIdx.x;
  if (i >= total) return;
  int c = (int)(i / LP), p = (int)(i % LP), l = p - pad;
  y[i] = (l >= 0 && l < L) ? f2bf(x[(size_t)c * L + l]) : (unsigned short)0;
}

// pack fp32 -> bf16 with generic strides: dst[r*Cc+c] = bf16(src[base + r*sr + c*sc])
__global__ void k_pack_bf16(const float* __restrict__ src, unsigned short* __restrict__ dst,
                            int Cc, long long sr, long long sc, long long base, size_t total)
{
  size_t i = (size_t)blockIdx.x * 256 + threadIdx.x;
  if (i >= total) return;
  long long r = (long long)(i / Cc), c = (long long)(i % Cc);
  dst[i] = f2bf(src[base + r * sr + c * sc]);
}

__global__ void k_bias_fill(float* __restrict__ Cm, const float* __restrict__ bias,
                            int N, int ldc, int per_row, size_t total)
{
  size_t i = (size_t)blockIdx.x * 256 + threadIdx.x;
  if (i >= total) return;
  int m = (int)(i / N), n = (int)(i % N);
  Cm[(size_t)m * ldc + n] = bias[per_row ? m : n];
}

__global__ void k_add(float* __restrict__ a, const float* __restrict__ b, size_t total)
{
  size_t i = (size_t)blockIdx.x * 256 + threadIdx.x;
  if (i < total) a[i] += b[i];
}

__global__ void k_attn_pool(const float* __restrict__ h, const float* __restrict__ sc,
                            float* __restrict__ out, int L, size_t total)
{
  size_t i = (size_t)blockIdx.x * 256 + threadIdx.x;
  if (i >= total) return;
  int W = L >> 1;
  int c = (int)(i / W), w = (int)(i % W);
  size_t base = (size_t)c * L + 2 * w;
  float s0 = sc[base], s1 = sc[base + 1];
  float m = fmaxf(s0, s1);
  float e0 = expf(s0 - m), e1 = expf(s1 - m);
  out[i] = (e0 * h[base] + e1 * h[base + 1]) / (e0 + e1);
}

// dst[R x C] = transpose: dst[r*C+c] = src[c*R + r]
__global__ void k_transpose(float* __restrict__ dst, const float* __restrict__ src,
                            int R, int C, size_t total)
{
  size_t i = (size_t)blockIdx.x * 256 + threadIdx.x;
  if (i >= total) return;
  int r = (int)(i / C), c = (int)(i % C);
  dst[i] = src[(size_t)c * R + r];
}

// mode: 0 copy, 1 relu, 2 gelu
__global__ void k_act_bf16(const float* __restrict__ x, unsigned short* __restrict__ y,
                           int mode, size_t total)
{
  size_t i = (size_t)blockIdx.x * 256 + threadIdx.x;
  if (i >= total) return;
  float v = x[i];
  if (mode == 1) v = fmaxf(v, 0.f);
  else if (mode == 2) v = gelu_exact(v);
  y[i] = f2bf(v);
}

__global__ void k_layernorm_bf16(const float* __restrict__ x, const float* __restrict__ g,
                                 const float* __restrict__ b, unsigned short* __restrict__ y,
                                 int C)
{
  int r = blockIdx.x;
  __shared__ float s1[256], s2[256];
  float a = 0.f, q = 0.f;
  for (int c = threadIdx.x; c < C; c += 256) {
    float v = x[(size_t)r * C + c]; a += v; q += v * v;
  }
  s1[threadIdx.x] = a; s2[threadIdx.x] = q; __syncthreads();
  for (int o = 128; o > 0; o >>= 1) {
    if (threadIdx.x < o) { s1[threadIdx.x] += s1[threadIdx.x + o]; s2[threadIdx.x] += s2[threadIdx.x + o]; }
    __syncthreads();
  }
  float m  = s1[0] / (float)C;
  float rs = rsqrtf(s2[0] / (float)C - m * m + 1e-5f);
  for (int c = threadIdx.x; c < C; c += 256)
    y[(size_t)r * C + c] = f2bf((x[(size_t)r * C + c] - m) * rs * g[c] + b[c]);
}

// Relative position basis: exponential + central-mask features (+signed copies)
__global__ void k_pos_ef(float* __restrict__ basis, int L, size_t total)
{
  size_t i = (size_t)blockIdx.x * 256 + threadIdx.x;
  if (i >= total) return;
  int r = (int)(i / 32), j = (int)(i % 32);
  float pos = (float)(r - (L - 1));
  float ap = fabsf(pos);
  float s = (pos > 0.f) ? 1.f : ((pos < 0.f) ? -1.f : 0.f);
  float hl = exp2f(3.0f + (log2f((float)L) - 3.0f) * (float)j / 31.0f);
  float fe = exp2f(-ap / hl);
  float fm = (exp2f((float)(j + 1)) > ap) ? 1.f : 0.f;
  float* row = basis + (size_t)r * 192;
  row[j] = fe; row[32 + j] = s * fe; row[64 + j] = fm; row[96 + j] = s * fm;
}

// Gamma features: compute raw pdf, column max, normalize (+signed copy)
__global__ void k_pos_gamma(float* __restrict__ basis, float* __restrict__ raw, int L)
{
  int j = blockIdx.x;                // 0..31
  int L2 = 2 * L - 1;
  float sl = (float)L;
  float stddev = sl / 64.0f;
  float mean = sl / 32.0f + (sl - sl / 32.0f) * (float)j / 31.0f;
  float conc = (mean / stddev) * (mean / stddev);
  float rate = mean / (stddev * stddev);
  float lgc = lgammaf(conc);
  float clr = conc * logf(rate);
  __shared__ float red[256];
  float mx = -1e30f;
  for (int r = threadIdx.x; r < L2; r += 256) {
    float ap = fabsf((float)(r - (L - 1)));
    float lx = logf(fmaxf(ap, 1e-8f));
    float fg = expf(clr + (conc - 1.0f) * lx - rate * ap - lgc) + 1e-8f;
    raw[(size_t)r * 32 + j] = fg;
    mx = fmaxf(mx, fg);
  }
  red[threadIdx.x] = mx; __syncthreads();
  for (int o = 128; o > 0; o >>= 1) {
    if (threadIdx.x < o) red[threadIdx.x] = fmaxf(red[threadIdx.x], red[threadIdx.x + o]);
    __syncthreads();
  }
  float inv = 1.0f / red[0];
  for (int r = threadIdx.x; r < L2; r += 256) {
    float v = raw[(size_t)r * 32 + j] * inv;
    float pos = (float)(r - (L - 1));
    float s = (pos > 0.f) ? 1.f : ((pos < 0.f) ? -1.f : 0.f);
    basis[(size_t)r * 192 + 128 + j] = v;
    basis[(size_t)r * 192 + 160 + j] = s * v;
  }
}

__global__ void k_compute_R(const float* __restrict__ basis, const float* __restrict__ wr,
                            float* __restrict__ R, size_t total)
{
  size_t i = (size_t)blockIdx.x * 256 + threadIdx.x;
  if (i >= total) return;
  int r = (int)(i / 64), d = (int)(i % 64);
  float acc = 0.f;
  for (int f = 0; f < 192; ++f) acc += basis[(size_t)r * 192 + f] * wr[f * 64 + d];
  R[i] = acc;
}

__global__ void k_t3(const float* __restrict__ Kmat, const float* __restrict__ u,
                     float* __restrict__ t3, size_t total)
{
  size_t i = (size_t)blockIdx.x * 256 + threadIdx.x;
  if (i >= total) return;
  int h = (int)(i / 768), kpos = (int)(i % 768);
  float acc = 0.f;
  for (int d = 0; d < 64; ++d) acc += u[h * 64 + d] * Kmat[(size_t)kpos * 512 + h * 64 + d];
  t3[i] = acc;
}

__global__ void k_t4(const float* __restrict__ R, const float* __restrict__ v,
                     float* __restrict__ t4)
{
  int h = threadIdx.x;
  if (h >= 8) return;
  float acc = 0.f;
  for (int d = 0; d < 64; ++d) acc += v[h * 64 + d] * R[d];  // rel_shift collapses to R[0]
  t4[h] = acc;
}

// logits = (t1 + rel_shift(t2) + t3 + t4) * 0.125 ; softmax over k ; bf16 out
__global__ void k_attn_softmax(const float* __restrict__ t1, const float* __restrict__ t2,
                               const float* __restrict__ t3, const float* __restrict__ t4,
                               unsigned short* __restrict__ attnb)
{
  int hq = blockIdx.x;
  int h = hq / 768, q = hq % 768;
  __shared__ float red[256];
  float vals[3];
  float mx = -1e30f;
  #pragma unroll
  for (int it = 0; it < 3; ++it) {
    int k = threadIdx.x + it * 256;
    // exact reference rel_shift: i=(q+1)*L+k; qs=i/2L; j=i%2L; j==0 -> 0 else t2[qs][j-1]
    size_t idx = (size_t)(q + 1) * 768 + k;
    int qs = (int)(idx / 1536), j = (int)(idx % 1536);
    float sh = (j == 0) ? 0.f : t2[((size_t)h * 768 + qs) * 1535 + (j - 1)];
    float v = (t1[((size_t)h * 768 + q) * 768 + k] + sh + t3[h * 768 + k] + t4[h]) * 0.125f;
    vals[it] = v; mx = fmaxf(mx, v);
  }
  red[threadIdx.x] = mx; __syncthreads();
  for (int o = 128; o > 0; o >>= 1) {
    if (threadIdx.x < o) red[threadIdx.x] = fmaxf(red[threadIdx.x], red[threadIdx.x + o]);
    __syncthreads();
  }
  mx = red[0]; __syncthreads();
  float sum = 0.f;
  #pragma unroll
  for (int it = 0; it < 3; ++it) { vals[it] = expf(vals[it] - mx); sum += vals[it]; }
  red[threadIdx.x] = sum; __syncthreads();
  for (int o = 128; o > 0; o >>= 1) {
    if (threadIdx.x < o) red[threadIdx.x] += red[threadIdx.x + o];
    __syncthreads();
  }
  float inv = 1.0f / red[0];
  #pragma unroll
  for (int it = 0; it < 3; ++it)
    attnb[((size_t)h * 768 + q) * 768 + threadIdx.x + it * 256] = f2bf(vals[it] * inv);
}

__global__ void k_softplus(const float* __restrict__ x, float* __restrict__ y, size_t total)
{
  size_t i = (size_t)blockIdx.x * 256 + threadIdx.x;
  if (i >= total) return;
  float v = x[i];
  y[i] = (v > 20.f) ? v : log1pf(expf(v));
}

// ---------------------------------------------------------------------------
// Host orchestration
// ---------------------------------------------------------------------------
static inline dim3 g1(size_t n) { return dim3((unsigned int)((n + 255) / 256)); }

static inline void gemm(hipStream_t s, const unsigned short* A, int lda,
                        const unsigned short* B, int ldb, float* C, int ldc,
                        int M, int N, int K, int accum)
{
  dim3 grid((unsigned)((N + BN - 1) / BN), (unsigned)((M + BM - 1) / BM));
  k_gemm_wmma<<<grid, 256, 0, s>>>(A, lda, B, ldb, C, ldc, M, N, K, accum);
}

extern "C" void kernel_launch(void* const* d_in, const int* in_sizes, int n_in,
                              void* d_out, int out_size, void* d_ws, size_t ws_size,
                              hipStream_t stream)
{
  (void)in_sizes; (void)n_in; (void)out_size; (void)ws_size;
  int ii = 0;
  auto IN = [&]() { return (const float*)d_in[ii++]; };

  const float* x_in = IN();
  struct { const float *conv_w, *conv_b, *bn_g, *bn_b, *rconv_w, *rconv_b, *pool_w; } st;
  st.conv_w = IN(); st.conv_b = IN(); st.bn_g = IN(); st.bn_b = IN();
  st.rconv_w = IN(); st.rconv_b = IN(); st.pool_w = IN();

  struct TowerP { const float *bn1_g,*bn1_b,*conv1_w,*conv1_b,*bn2_g,*bn2_b,*conv2_w,*conv2_b,*pool_w; } tw[6];
  for (int i = 0; i < 6; ++i) {
    tw[i].bn1_g = IN(); tw[i].bn1_b = IN(); tw[i].conv1_w = IN(); tw[i].conv1_b = IN();
    tw[i].bn2_g = IN(); tw[i].bn2_b = IN(); tw[i].conv2_w = IN(); tw[i].conv2_b = IN();
    tw[i].pool_w = IN();
  }
  struct LayerP { const float *ln1_g,*ln1_b,*q_w,*q_b,*k_w,*k_b,*v_w,*v_b,*o_w,*o_b,*wr,*u,*v,
                              *ln2_g,*ln2_b,*ff1_w,*ff1_b,*ff2_w,*ff2_b; } ly[11];
  for (int i = 0; i < 11; ++i) {
    ly[i].ln1_g = IN(); ly[i].ln1_b = IN();
    ly[i].q_w = IN(); ly[i].q_b = IN(); ly[i].k_w = IN(); ly[i].k_b = IN();
    ly[i].v_w = IN(); ly[i].v_b = IN(); ly[i].o_w = IN(); ly[i].o_b = IN();
    ly[i].wr = IN(); ly[i].u = IN(); ly[i].v = IN();
    ly[i].ln2_g = IN(); ly[i].ln2_b = IN();
    ly[i].ff1_w = IN(); ly[i].ff1_b = IN(); ly[i].ff2_w = IN(); ly[i].ff2_b = IN();
  }
  struct { const float *bn_g,*bn_b,*conv_w,*conv_b; } pw;
  pw.bn_g = IN(); pw.bn_b = IN(); pw.conv_w = IN(); pw.conv_b = IN();
  struct { const float *conv_w,*conv_b; } hd;
  hd.conv_w = IN(); hd.conv_b = IN();

  // ---- workspace bump allocator ----
  char* ws = (char*)d_ws;
  size_t cur = 0;
  auto alloc_f = [&](size_t n) { float* p = (float*)(ws + cur); cur += ((n * 4 + 255) / 256) * 256; return p; };
  auto alloc_h = [&](size_t n) { unsigned short* p = (unsigned short*)(ws + cur); cur += ((n * 2 + 255) / 256) * 256; return p; };

  const size_t ACT = 75497472ull;               // max C*L activation elements
  float* h32 = alloc_f(ACT);
  float* t32 = alloc_f(ACT);
  float* a32 = alloc_f(ACT);
  unsigned short* bfA = alloc_h(75500544ull);   // bf16 padded activations
  unsigned short* wbf = alloc_h(5313ull * 3072);// bf16 weight scratch (max: head)
  float* mean = alloc_f(4096);
  float* rvar = alloc_f(4096);
  float* basis = alloc_f(1535ull * 192);
  float* fgraw = alloc_f(1535ull * 32);
  float* Rm = alloc_f(1535ull * 64);
  unsigned short* RT = alloc_h(64ull * 1535);
  unsigned short* xln = alloc_h(768ull * 1536);
  float* q32 = alloc_f(768ull * 512);
  float* k32 = alloc_f(768ull * 512);
  float* v32 = alloc_f(768ull * 1536);
  unsigned short* qb = alloc_h(768ull * 512);
  unsigned short* kT = alloc_h(512ull * 768);
  unsigned short* vb = alloc_h(768ull * 1536);
  float* t1b = alloc_f(8ull * 768 * 768);
  float* t2b = alloc_f(8ull * 768 * 1535);
  unsigned short* attnb = alloc_h(8ull * 768 * 768);
  float* o32 = alloc_f(768ull * 1536);
  unsigned short* obf = alloc_h(768ull * 1536);
  float* m32 = alloc_f(768ull * 3072);
  unsigned short* mbf = alloc_h(768ull * 3072);
  float* t3v = alloc_f(8ull * 768);
  float* t4v = alloc_f(8);

  // =================== STEM ===================
  int L = 98304;
  k_stem_conv15<<<g1((size_t)768 * L), 256, 0, stream>>>(x_in, st.conv_w, st.conv_b, h32, L);

  int LP = L + 4;
  k_bn_stats<<<768, 256, 0, stream>>>(h32, L, mean, rvar);
  k_bn_gelu_pad<<<g1((size_t)768 * LP), 256, 0, stream>>>(h32, mean, rvar, st.bn_g, st.bn_b,
                                                          bfA, L, LP, 2, (size_t)768 * LP);
  k_pack_bf16<<<g1(768ull * 768), 256, 0, stream>>>(st.rconv_w, wbf, 768, 768, 1, 0, 768ull * 768);
  k_bias_fill<<<g1((size_t)768 * L), 256, 0, stream>>>(t32, st.rconv_b, L, L, 1, (size_t)768 * L);
  gemm(stream, wbf, 768, bfA + 2, LP, t32, L, 768, L, 768, 1);
  k_add<<<g1((size_t)768 * L), 256, 0, stream>>>(h32, t32, (size_t)768 * L);

  // attention pool (scores = pool_w^T @ h)
  k_pack_bf16<<<g1(768ull * 768), 256, 0, stream>>>(st.pool_w, wbf, 768, 1, 768, 0, 768ull * 768);
  k_cvt_pad<<<g1((size_t)768 * LP), 256, 0, stream>>>(h32, bfA, L, LP, 2, (size_t)768 * LP);
  gemm(stream, wbf, 768, bfA + 2, LP, a32, L, 768, L, 768, 0);
  k_attn_pool<<<g1((size_t)768 * (L / 2)), 256, 0, stream>>>(h32, a32, t32, L, (size_t)768 * (L / 2));
  { float* tmp = h32; h32 = t32; t32 = tmp; }
  L /= 2;

  // =================== TOWER ===================
  const int ch[7] = {768, 862, 968, 1086, 1219, 1368, 1536};
  for (int i = 0; i < 6; ++i) {
    const TowerP& T = tw[i];
    int Ci = ch[i], Co = ch[i + 1];
    LP = L + 4;
    // conv_block: BN -> GELU -> conv(k=5,pad=2) as 5 shifted accumulating GEMMs
    k_bn_stats<<<Ci, 256, 0, stream>>>(h32, L, mean, rvar);
    k_bn_gelu_pad<<<g1((size_t)Ci * LP), 256, 0, stream>>>(h32, mean, rvar, T.bn1_g, T.bn1_b,
                                                           bfA, L, LP, 2, (size_t)Ci * LP);
    k_bias_fill<<<g1((size_t)Co * L), 256, 0, stream>>>(t32, T.conv1_b, L, L, 1, (size_t)Co * L);
    for (int t = 0; t < 5; ++t) {
      k_pack_bf16<<<g1((size_t)Co * Ci), 256, 0, stream>>>(T.conv1_w, wbf, Ci,
                                                           (long long)Ci * 5, 5, t, (size_t)Co * Ci);
      gemm(stream, wbf, Ci, bfA + t, LP, t32, L, Co, L, Ci, 1);
    }
    // residual 1x1 conv_block
    k_bn_stats<<<Co, 256, 0, stream>>>(t32, L, mean, rvar);
    k_bn_gelu_pad<<<g1((size_t)Co * LP), 256, 0, stream>>>(t32, mean, rvar, T.bn2_g, T.bn2_b,
                                                           bfA, L, LP, 2, (size_t)Co * LP);
    k_pack_bf16<<<g1((size_t)Co * Co), 256, 0, stream>>>(T.conv2_w, wbf, Co, Co, 1, 0, (size_t)Co * Co);
    k_bias_fill<<<g1((size_t)Co * L), 256, 0, stream>>>(a32, T.conv2_b, L, L, 1, (size_t)Co * L);
    gemm(stream, wbf, Co, bfA + 2, LP, a32, L, Co, L, Co, 1);
    k_add<<<g1((size_t)Co * L), 256, 0, stream>>>(t32, a32, (size_t)Co * L);
    // attention pool
    k_pack_bf16<<<g1((size_t)Co * Co), 256, 0, stream>>>(T.pool_w, wbf, Co, 1, Co, 0, (size_t)Co * Co);
    k_cvt_pad<<<g1((size_t)Co * LP), 256, 0, stream>>>(t32, bfA, L, LP, 2, (size_t)Co * LP);
    gemm(stream, wbf, Co, bfA + 2, LP, a32, L, Co, L, Co, 0);
    k_attn_pool<<<g1((size_t)Co * (L / 2)), 256, 0, stream>>>(t32, a32, h32, L, (size_t)Co * (L / 2));
    L /= 2;
  }
  // h32: [1536][768] -> transpose to [768][1536] residual stream in a32
  k_transpose<<<g1(768ull * 1536), 256, 0, stream>>>(a32, h32, 768, 1536, 768ull * 1536);
  float* hs = a32;

  // =================== POSITION BASIS (L=768) ===================
  k_pos_ef<<<g1(1535ull * 32), 256, 0, stream>>>(basis, 768, 1535ull * 32);
  k_pos_gamma<<<32, 256, 0, stream>>>(basis, fgraw, 768);

  // =================== TRANSFORMER (11 layers) ===================
  for (int li = 0; li < 11; ++li) {
    const LayerP& P = ly[li];
    k_compute_R<<<g1(1535ull * 64), 256, 0, stream>>>(basis, P.wr, Rm, 1535ull * 64);
    k_pack_bf16<<<g1(64ull * 1535), 256, 0, stream>>>(Rm, RT, 1535, 1, 64, 0, 64ull * 1535); // R^T

    k_layernorm_bf16<<<768, 256, 0, stream>>>(hs, P.ln1_g, P.ln1_b, xln, 1536);

    // Q, K, V projections
    k_pack_bf16<<<g1(1536ull * 512), 256, 0, stream>>>(P.q_w, wbf, 512, 512, 1, 0, 1536ull * 512);
    k_bias_fill<<<g1(768ull * 512), 256, 0, stream>>>(q32, P.q_b, 512, 512, 0, 768ull * 512);
    gemm(stream, xln, 1536, wbf, 512, q32, 512, 768, 512, 1536, 1);

    k_pack_bf16<<<g1(1536ull * 512), 256, 0, stream>>>(P.k_w, wbf, 512, 512, 1, 0, 1536ull * 512);
    k_bias_fill<<<g1(768ull * 512), 256, 0, stream>>>(k32, P.k_b, 512, 512, 0, 768ull * 512);
    gemm(stream, xln, 1536, wbf, 512, k32, 512, 768, 512, 1536, 1);

    k_pack_bf16<<<g1(1536ull * 1536), 256, 0, stream>>>(P.v_w, wbf, 1536, 1536, 1, 0, 1536ull * 1536);
    k_bias_fill<<<g1(768ull * 1536), 256, 0, stream>>>(v32, P.v_b, 1536, 1536, 0, 768ull * 1536);
    gemm(stream, xln, 1536, wbf, 1536, v32, 1536, 768, 1536, 1536, 1);

    k_act_bf16<<<g1(768ull * 512), 256, 0, stream>>>(q32, qb, 0, 768ull * 512);
    k_pack_bf16<<<g1(512ull * 768), 256, 0, stream>>>(k32, kT, 768, 1, 512, 0, 512ull * 768); // K^T
    k_act_bf16<<<g1(768ull * 1536), 256, 0, stream>>>(v32, vb, 0, 768ull * 1536);

    k_t3<<<g1(8ull * 768), 256, 0, stream>>>(k32, P.u, t3v, 8ull * 768);
    k_t4<<<1, 8, 0, stream>>>(Rm, P.v, t4v);

    for (int h = 0; h < 8; ++h) {
      gemm(stream, qb + h * 64, 512, kT + (size_t)h * 64 * 768, 768,
           t1b + (size_t)h * 768 * 768, 768, 768, 768, 64, 0);                 // Q·K^T
      gemm(stream, qb + h * 64, 512, RT, 1535,
           t2b + (size_t)h * 768 * 1535, 1535, 768, 1535, 64, 0);              // Q·R^T
    }
    k_attn_softmax<<<8 * 768, 256, 0, stream>>>(t1b, t2b, t3v, t4v, attnb);
    for (int h = 0; h < 8; ++h)
      gemm(stream, attnb + (size_t)h * 768 * 768, 768, vb + h * 192, 1536,
           o32 + h * 192, 1536, 768, 192, 768, 0);                             // attn·V

    // output projection + residual
    k_act_bf16<<<g1(768ull * 1536), 256, 0, stream>>>(o32, obf, 0, 768ull * 1536);
    k_pack_bf16<<<g1(1536ull * 1536), 256, 0, stream>>>(P.o_w, wbf, 1536, 1536, 1, 0, 1536ull * 1536);
    k_bias_fill<<<g1(768ull * 1536), 256, 0, stream>>>(t32, P.o_b, 1536, 1536, 0, 768ull * 1536);
    gemm(stream, obf, 1536, wbf, 1536, t32, 1536, 768, 1536, 1536, 1);
    k_add<<<g1(768ull * 1536), 256, 0, stream>>>(hs, t32, 768ull * 1536);

    // FFN + residual
    k_layernorm_bf16<<<768, 256, 0, stream>>>(hs, P.ln2_g, P.ln2_b, xln, 1536);
    k_pack_bf16<<<g1(1536ull * 3072), 256, 0, stream>>>(P.ff1_w, wbf, 3072, 3072, 1, 0, 1536ull * 3072);
    k_bias_fill<<<g1(768ull * 3072), 256, 0, stream>>>(m32, P.ff1_b, 3072, 3072, 0, 768ull * 3072);
    gemm(stream, xln, 1536, wbf, 3072, m32, 3072, 768, 3072, 1536, 1);
    k_act_bf16<<<g1(768ull * 3072), 256, 0, stream>>>(m32, mbf, 1, 768ull * 3072);
    k_pack_bf16<<<g1(3072ull * 1536), 256, 0, stream>>>(P.ff2_w, wbf, 1536, 1536, 1, 0, 3072ull * 1536);
    k_bias_fill<<<g1(768ull * 1536), 256, 0, stream>>>(t32, P.ff2_b, 1536, 1536, 0, 768ull * 1536);
    gemm(stream, mbf, 3072, wbf, 1536, t32, 1536, 768, 1536, 3072, 1);
    k_add<<<g1(768ull * 1536), 256, 0, stream>>>(hs, t32, 768ull * 1536);
  }

  // =================== HEAD ===================
  // crop rows [320,448) of [768][1536], transpose to [1536][128]
  k_transpose<<<g1(1536ull * 128), 256, 0, stream>>>(h32, hs + 320 * 1536, 1536, 128, 1536ull * 128);
  k_bn_stats<<<1536, 256, 0, stream>>>(h32, 128, mean, rvar);
  k_bn_gelu_pad<<<g1(1536ull * 128), 256, 0, stream>>>(h32, mean, rvar, pw.bn_g, pw.bn_b,
                                                       bfA, 128, 128, 0, 1536ull * 128);
  k_pack_bf16<<<g1(3072ull * 1536), 256, 0, stream>>>(pw.conv_w, wbf, 1536, 1536, 1, 0, 3072ull * 1536);
  k_bias_fill<<<g1(3072ull * 128), 256, 0, stream>>>(t32, pw.conv_b, 128, 128, 1, 3072ull * 128);
  gemm(stream, wbf, 1536, bfA, 128, t32, 128, 3072, 128, 1536, 1);
  k_act_bf16<<<g1(3072ull * 128), 256, 0, stream>>>(t32, bfA, 2, 3072ull * 128);   // gelu -> bf16
  k_pack_bf16<<<g1(5313ull * 3072), 256, 0, stream>>>(hd.conv_w, wbf, 3072, 3072, 1, 0, 5313ull * 3072);
  k_bias_fill<<<g1(5313ull * 128), 256, 0, stream>>>(h32, hd.conv_b, 128, 128, 1, 5313ull * 128);
  gemm(stream, wbf, 3072, bfA, 128, h32, 128, 5313, 128, 3072, 1);
  k_softplus<<<g1(5313ull * 128), 256, 0, stream>>>(h32, (float*)d_out, 5313ull * 128);
}
